// TimeAwareAttention_3212635537839
// MI455X (gfx1250) — matrix operand order repaired and checked by reading, not compiled, not run on hardware
//
#include <hip/hip_runtime.h>
#include <hip/hip_bf16.h>
#include <math.h>

#define B_ 32
#define S_ 2048
#define H_ 1024

typedef __attribute__((ext_vector_type(16))) __bf16 v16bf;
typedef __attribute__((ext_vector_type(8)))  float  v8f;
typedef __attribute__((ext_vector_type(4)))  float  v4f;
typedef __attribute__((ext_vector_type(2)))  unsigned int v2u;

#if __has_builtin(__builtin_amdgcn_tanhf)
#define fast_tanh(x) __builtin_amdgcn_tanhf(x)
#else
#define fast_tanh(x) tanhf(x)
#endif

static __device__ __forceinline__ unsigned short f2bf(float f) {
    unsigned int u = __float_as_uint(f);
    unsigned int r = u + 0x7FFFu + ((u >> 16) & 1u);   // round-to-nearest-even
    return (unsigned short)(r >> 16);
}

// ---------------------------------------------------------------------------
// Kernel 1: swizzle Wk (fp32 [H,H] row-major) into bf16 WMMA-B layout:
//   Bsw[((kt*64 + nt)*32 + lane)*16 + j]  (stored as uint32 pairs)
//   lane<16 : K = kt*32 + j       , N = nt*16 + (lane&15)
//   lane>=16: K = kt*32 + 16 + j  , N = nt*16 + (lane&15)
// ---------------------------------------------------------------------------
__global__ void swizzle_wk(const float* __restrict__ Wk, unsigned int* __restrict__ Bsw) {
    int id = blockIdx.x * blockDim.x + threadIdx.x;   // 0 .. 524287 (bf16 pairs)
    int j2 = id & 7;
    int L  = (id >> 3) & 31;
    int nt = (id >> 8) & 63;
    int kt = id >> 14;
    int n  = nt * 16 + (L & 15);
    int k0 = kt * 32 + ((L >> 4) << 4) + 2 * j2;
    unsigned int lo = f2bf(Wk[(size_t)k0 * H_ + n]);
    unsigned int hi = f2bf(Wk[(size_t)(k0 + 1) * H_ + n]);
    Bsw[id] = lo | (hi << 16);
}

// ---------------------------------------------------------------------------
// Kernel 2: q[b,h] = query[b,:] @ Wq[:,h] + bq[h]   (tiny)
// ---------------------------------------------------------------------------
__global__ void compute_q(const float* __restrict__ query, const float* __restrict__ Wq,
                          const float* __restrict__ bq, float* __restrict__ qbuf) {
    int id = blockIdx.x * blockDim.x + threadIdx.x;   // 0 .. 32767
    int b = id >> 10, h = id & 1023;
    float acc = bq[h];
    const float* qr = query + (size_t)b * H_;
    for (int k = 0; k < H_; ++k) acc = fmaf(qr[k], Wq[(size_t)k * H_ + h], acc);
    qbuf[id] = acc;
}

// ---------------------------------------------------------------------------
// Kernel 3: fused  scores[b,s] = sum_h tanh(seq@Wk + bk + q + tf@Wt + bt)*Ws + bs
// Block = 64-row M-tile (4 WMMA row-tiles), 8 wave32.
// Wave w owns nt in [8w, 8w+8), processed as 4 subs of NT=2 with MT=4 register
// blocking: per K-step, 4 A frags (LDS) + 2 B frags (L2) -> 8 WMMAs.
// A tile is stored K-step-major in LDS (m-tiles at +1KB strides) so all DS
// accesses use 16-bit immediate offsets -> no per-iteration address VALU ->
// no WMMA->VALU WAR hazard NOPs in the hot loop.
// ---------------------------------------------------------------------------
__global__ __launch_bounds__(256) void fused_scores(
    const float* __restrict__ seq, const float* __restrict__ tfeat,
    const unsigned int* __restrict__ Bsw, const float* __restrict__ qbuf,
    const float* __restrict__ bk, const float* __restrict__ Wt,
    const float* __restrict__ bt, const float* __restrict__ Ws,
    const float* __restrict__ bs, float* __restrict__ scores)
{
    extern __shared__ unsigned int smem[];
    unsigned int* ldsA  = smem;                       // 32768 u32 = 128 KB A tile
    float*        ltf   = (float*)(smem + 32768);     // 128 floats (64 rows x 2)
    float*        spart = ltf + 128;                  // 8 waves x 64 rows

    const int tid = threadIdx.x;
    const int mt  = blockIdx.x;                       // 0..1023
    const int b   = mt >> 5;                          // 32 blocks per batch row
    const int s0  = (mt & 31) << 6;
    const size_t rowbase = (size_t)b * S_ + s0;

    // --- cooperative load of 64 x 1024 fp32 A rows -> bf16 WMMA-A layout ----
    // LDS u32 index: kt*1024 + mtile*256 + lane*8 + pos/2
    {
        const int col = tid * 4;
        const int kt  = col >> 5;
        const int kp  = col & 31;
        const int g   = kp >> 3;
        const int lhalf = (g >= 2) ? 16 : 0;
        const int pos = kp & 15;
        for (int r = 0; r < 64; ++r) {
            const v4f v = __builtin_nontemporal_load(
                (const v4f*)(seq + (rowbase + r) * H_ + col));
            int lane = (r & 15) + lhalf;
            int idx  = kt * 1024 + (r >> 4) * 256 + lane * 8 + (pos >> 1);
            v2u p;
            p.x = (unsigned int)f2bf(v.x) | ((unsigned int)f2bf(v.y) << 16);
            p.y = (unsigned int)f2bf(v.z) | ((unsigned int)f2bf(v.w) << 16);
            *(v2u*)&ldsA[idx] = p;
        }
    }
    if (tid < 128) ltf[tid] = tfeat[rowbase * 2 + tid];   // row = tid/2, feat = tid&1
    __syncthreads();

    const int lane   = tid & 31;
    const int wave   = tid >> 5;
    const int ncol   = lane & 15;
    const int rowoff = (lane < 16) ? 0 : 8;
    const unsigned int* ap = ldsA + lane * 8;

    float sacc[4][8];
#pragma unroll
    for (int m = 0; m < 4; ++m)
#pragma unroll
        for (int j = 0; j < 8; ++j) sacc[m][j] = 0.0f;

    for (int sub = 0; sub < 4; ++sub) {
        const int nt0 = wave * 8 + sub * 2;
        v8f acc[4][2];
#pragma unroll
        for (int m = 0; m < 4; ++m)
#pragma unroll
            for (int n = 0; n < 2; ++n)
                acc[m][n] = (v8f){0.f,0.f,0.f,0.f,0.f,0.f,0.f,0.f};

        const unsigned int* bp0 = Bsw + ((size_t)(nt0 * 32 + lane) * 8);
        const unsigned int* bp1 = Bsw + ((size_t)((nt0 + 1) * 32 + lane) * 8);

        for (int kt = 0; kt < 32; ++kt) {
            const unsigned int* akt = ap + kt * 1024;
            v16bf a0 = *(const v16bf*)(akt);
            v16bf a1 = *(const v16bf*)(akt + 256);
            v16bf a2 = *(const v16bf*)(akt + 512);
            v16bf a3 = *(const v16bf*)(akt + 768);
            v16bf b0 = *(const v16bf*)(bp0 + (size_t)kt * 16384);
            v16bf b1 = *(const v16bf*)(bp1 + (size_t)kt * 16384);
            acc[0][0] = __builtin_amdgcn_wmma_f32_16x16x32_bf16(false, a0, false, b0, (short)0, acc[0][0], false, false);
            acc[1][0] = __builtin_amdgcn_wmma_f32_16x16x32_bf16(false, a1, false, b0, (short)0, acc[1][0], false, false);
            acc[2][0] = __builtin_amdgcn_wmma_f32_16x16x32_bf16(false, a2, false, b0, (short)0, acc[2][0], false, false);
            acc[3][0] = __builtin_amdgcn_wmma_f32_16x16x32_bf16(false, a3, false, b0, (short)0, acc[3][0], false, false);
            acc[0][1] = __builtin_amdgcn_wmma_f32_16x16x32_bf16(false, a0, false, b1, (short)0, acc[0][1], false, false);
            acc[1][1] = __builtin_amdgcn_wmma_f32_16x16x32_bf16(false, a1, false, b1, (short)0, acc[1][1], false, false);
            acc[2][1] = __builtin_amdgcn_wmma_f32_16x16x32_bf16(false, a2, false, b1, (short)0, acc[2][1], false, false);
            acc[3][1] = __builtin_amdgcn_wmma_f32_16x16x32_bf16(false, a3, false, b1, (short)0, acc[3][1], false, false);
        }

        // epilogue for the two N-tiles of this sub
        const int n0 = nt0 * 16 + ncol;
        const int n1 = n0 + 16;
        const float wt00 = Wt[n0],       wt01 = Wt[n1];
        const float wt10 = Wt[H_ + n0],  wt11 = Wt[H_ + n1];
        const float bse0 = bk[n0] + bt[n0] + qbuf[(size_t)b * H_ + n0];
        const float bse1 = bk[n1] + bt[n1] + qbuf[(size_t)b * H_ + n1];
        const float ws0  = Ws[n0],       ws1  = Ws[n1];
#pragma unroll
        for (int m = 0; m < 4; ++m) {
#pragma unroll
            for (int j = 0; j < 8; ++j) {
                const int row = m * 16 + rowoff + j;
                const float t0 = ltf[2 * row];
                const float t1 = ltf[2 * row + 1];
                float v0 = acc[m][0][j] + bse0 + t0 * wt00 + t1 * wt10;
                float v1 = acc[m][1][j] + bse1 + t0 * wt01 + t1 * wt11;
                sacc[m][j] += fast_tanh(v0) * ws0 + fast_tanh(v1) * ws1;
            }
        }
    }

    // reduce across the 16 lanes of each half (xor masks < 16 stay in-half)
#pragma unroll
    for (int msk = 8; msk >= 1; msk >>= 1)
#pragma unroll
        for (int m = 0; m < 4; ++m)
#pragma unroll
            for (int j = 0; j < 8; ++j)
                sacc[m][j] += __shfl_xor(sacc[m][j], msk, 32);

    if (ncol == 0)
#pragma unroll
        for (int m = 0; m < 4; ++m)
#pragma unroll
            for (int j = 0; j < 8; ++j)
                spart[wave * 64 + m * 16 + rowoff + j] = sacc[m][j];
    __syncthreads();

    if (tid < 64) {
        float s = bs[0];
#pragma unroll
        for (int w = 0; w < 8; ++w) s += spart[w * 64 + tid];
        scores[rowbase + tid] = s;
    }
}

// ---------------------------------------------------------------------------
// Kernel 4: masked softmax over S per batch row
// ---------------------------------------------------------------------------
__global__ void softmax_mask(const float* __restrict__ scores,
                             const long long* __restrict__ lengths,
                             float* __restrict__ weights) {
    __shared__ float red[256];
    const int b = blockIdx.x, tid = threadIdx.x;
    const long long len = lengths[b];
    float v[8];
    float mx = -INFINITY;
#pragma unroll
    for (int i = 0; i < 8; ++i) {
        int s = tid + i * 256;
        float x = scores[(size_t)b * S_ + s];
        if ((long long)s >= len) x = -INFINITY;
        v[i] = x;
        mx = fmaxf(mx, x);
    }
    red[tid] = mx; __syncthreads();
    for (int off = 128; off > 0; off >>= 1) {
        if (tid < off) red[tid] = fmaxf(red[tid], red[tid + off]);
        __syncthreads();
    }
    mx = red[0]; __syncthreads();
    float sum = 0.f;
#pragma unroll
    for (int i = 0; i < 8; ++i) { float e = __expf(v[i] - mx); v[i] = e; sum += e; }
    red[tid] = sum; __syncthreads();
    for (int off = 128; off > 0; off >>= 1) {
        if (tid < off) red[tid] += red[tid + off];
        __syncthreads();
    }
    const float inv = 1.0f / red[0];
#pragma unroll
    for (int i = 0; i < 8; ++i) weights[(size_t)b * S_ + tid + i * 256] = v[i] * inv;
}

// ---------------------------------------------------------------------------
// Kernels 5/6: out[b,h] = sum_s w[b,s] * seq[b,s,h]  (deterministic partials)
// ---------------------------------------------------------------------------
__global__ void weighted_partial(const float* __restrict__ seq,
                                 const float* __restrict__ weights,
                                 float* __restrict__ partial) {
    const int h  = blockIdx.x * 128 + threadIdx.x;  // H/128 = 8 blocks in x
    const int sc = blockIdx.y;                      // 8 chunks of 256 s
    const int b  = blockIdx.z;
    const int sbase = sc * 256;
    const float* sp = seq + ((size_t)b * S_ + sbase) * H_ + h;
    const float* wp = weights + (size_t)b * S_ + sbase;
    float acc = 0.f;
    for (int s = 0; s < 256; ++s)
        acc = fmaf(wp[s], __builtin_nontemporal_load(sp + (size_t)s * H_), acc);
    partial[((size_t)sc * B_ + b) * H_ + h] = acc;
}

__global__ void reduce_out(const float* __restrict__ partial, float* __restrict__ out) {
    int id = blockIdx.x * blockDim.x + threadIdx.x;  // 0 .. 32767
    float s = 0.f;
#pragma unroll
    for (int sc = 0; sc < 8; ++sc) s += partial[(size_t)sc * B_ * H_ + id];
    out[id] = s;
}

// ---------------------------------------------------------------------------
extern "C" void kernel_launch(void* const* d_in, const int* in_sizes, int n_in,
                              void* d_out, int out_size, void* d_ws, size_t ws_size,
                              hipStream_t stream) {
    (void)in_sizes; (void)n_in; (void)out_size; (void)ws_size;
    const float*     seq     = (const float*)d_in[0];
    const float*     query   = (const float*)d_in[1];
    const long long* lengths = (const long long*)d_in[2];
    const float*     tfeat   = (const float*)d_in[3];
    const float*     Wq      = (const float*)d_in[4];
    const float*     bq      = (const float*)d_in[5];
    const float*     Wk      = (const float*)d_in[6];
    const float*     bk      = (const float*)d_in[7];
    const float*     Wt      = (const float*)d_in[8];
    const float*     bt      = (const float*)d_in[9];
    const float*     Ws      = (const float*)d_in[10];
    const float*     bs      = (const float*)d_in[11];

    char* ws = (char*)d_ws;
    float*        qbuf    = (float*)(ws);                         // 128 KB
    unsigned int* Bsw     = (unsigned int*)(ws + (128u << 10));   // 2 MB
    float*        scores  = (float*)(ws + (128u << 10) + (2u << 20));
    float*        weights = (float*)(ws + (128u << 10) + (2u << 20) + (256u << 10));
    float*        partial = (float*)(ws + (128u << 10) + (2u << 20) + (512u << 10));

    const size_t lds_bytes = 32768u * 4u + 128u * 4u + 512u * 4u;  // A + tf + spart

    hipLaunchKernelGGL(swizzle_wk,       dim3(2048),     dim3(256), 0, stream, Wk, Bsw);
    hipLaunchKernelGGL(compute_q,        dim3(128),      dim3(256), 0, stream, query, Wq, bq, qbuf);
    hipLaunchKernelGGL(fused_scores,     dim3(1024),     dim3(256), lds_bytes, stream,
                       seq, tfeat, Bsw, qbuf, bk, Wt, bt, Ws, bs, scores);
    hipLaunchKernelGGL(softmax_mask,     dim3(32),       dim3(256), 0, stream, scores, lengths, weights);
    hipLaunchKernelGGL(weighted_partial, dim3(8, 8, 32), dim3(128), 0, stream, seq, weights, partial);
    hipLaunchKernelGGL(reduce_out,       dim3(128),      dim3(256), 0, stream, partial, (float*)d_out);
}
